// local_policy_att_50311246905511
// MI455X (gfx1250) — matrix operand order repaired and checked by reading, not compiled, not run on hardware
//
#include <hip/hip_runtime.h>
#include <hip/hip_bf16.h>

// Sizes from the reference
#define BB 32
#define MM 64
#define NN 5000
#define LL 100     // local_size (top-k)
#define LP 128     // L padded to multiple of 32 for WMMA K
#define HH 8
#define DD 16
#define EE 128
#define ROWS (BB * MM)   // 2048

typedef __attribute__((ext_vector_type(16))) _Float16 v16h;
typedef __attribute__((ext_vector_type(8)))  float    v8f;

// ---- monotonic float ordering (matches jax top_k tie-break on index) ----
__device__ __forceinline__ unsigned int f2ord(float f) {
    unsigned int u = __float_as_uint(f);
    return (u & 0x80000000u) ? ~u : (u | 0x80000000u);
}
__device__ __forceinline__ float ord2f(unsigned int o) {
    unsigned int u = (o & 0x80000000u) ? (o & 0x7FFFFFFFu) : ~o;
    return __uint_as_float(u);
}

// ---- CDNA5 async global->LDS copy (ASYNCcnt path, ISA 15.18.3 op 96) ----
// VDST holds the LDS byte offset; flat shared addresses carry the LDS offset
// in addr[31:0] (ISA 10.2 aperture mapping), so truncation is the offset.
__device__ __forceinline__ void async_g2l_b32(void* lds_ptr, const void* gptr) {
    unsigned loff = (unsigned)(unsigned long long)lds_ptr;
    unsigned long long ga = (unsigned long long)gptr;
    asm volatile("global_load_async_to_lds_b32 %0, %1, off"
                 :: "v"(loff), "v"(ga) : "memory");
}
__device__ __forceinline__ void wait_async0() {
    asm volatile("s_wait_asynccnt 0" ::: "memory");
}

// ---------------------------------------------------------------------------
// Kernel 0: zero the [B,M,N] output (scatter target)
// ---------------------------------------------------------------------------
__global__ void zero_kernel(float* __restrict__ out, int n) {
    int i = blockIdx.x * blockDim.x + threadIdx.x;
    int stride = gridDim.x * blockDim.x;
    for (; i < n; i += stride) out[i] = 0.0f;
}

// ---------------------------------------------------------------------------
// Kernel 1: shared query fusion: q = Wq@cur;  qk[h,e] = sum_d q[h*16+d]*Wk[h*16+d,e]
// ---------------------------------------------------------------------------
__global__ __launch_bounds__(128) void prep_qk_kernel(
    const float* __restrict__ Wq, const float* __restrict__ Wk,
    const float* __restrict__ cur, float* __restrict__ qk) {
    __shared__ float qS[HH * DD];
    int t = threadIdx.x;
    float acc = 0.0f;
    for (int e = 0; e < EE; ++e) acc += Wq[t * EE + e] * cur[e];
    qS[t] = acc;
    __syncthreads();
    for (int o = t; o < HH * EE; o += 128) {
        int h = o >> 7, e = o & 127;
        float s = 0.0f;
#pragma unroll
        for (int d = 0; d < DD; ++d) s += qS[h * DD + d] * Wk[(h * DD + d) * EE + e];
        qk[o] = s;
    }
}

// ---------------------------------------------------------------------------
// Kernel 2: per-row top-L via 64-bit radix select (8 histogram passes) +
// compaction + 128-wide bitonic sort. Keys = (ordered_float(d) << 32) | index,
// unique by construction -> exact kth pivot, jax-stable tie order.
// ---------------------------------------------------------------------------
__global__ __launch_bounds__(256) void topk_kernel(
    const float* __restrict__ dist, const float* __restrict__ theta,
    const float* __restrict__ ninf,
    float* __restrict__ sDist, float* __restrict__ sTheta,
    float* __restrict__ sMask, int* __restrict__ sIdx) {
    __shared__ unsigned long long keys[NN];   // 40 KB
    __shared__ unsigned int hist[256];
    __shared__ unsigned long long sel[128];
    __shared__ unsigned long long sh_prefix;
    __shared__ unsigned int sh_kth, selCnt;

    const int row = blockIdx.x;
    const int t = threadIdx.x;
    const size_t base = (size_t)row * NN;

    for (int n = t; n < NN; n += 256) {
        if (n + 2048 < NN) __builtin_prefetch(&dist[base + n + 2048], 0, 0);
        float d = dist[base + n] - ninf[base + n];   // masked -> +inf
        keys[n] = ((unsigned long long)f2ord(d) << 32) | (unsigned int)n;
    }
    if (t == 0) { sh_prefix = 0ull; sh_kth = LL; selCnt = 0u; }
    __syncthreads();

    // ---- radix select: find the 100th-smallest 64-bit key ----
    unsigned long long prefix = 0ull;
    for (int pass = 7; pass >= 0; --pass) {
        if (t < 256) hist[t] = 0u;
        __syncthreads();
        unsigned long long himask =
            (pass == 7) ? 0ull : (~0ull << (8 * (pass + 1)));
        for (int n = t; n < NN; n += 256) {
            unsigned long long kk = keys[n];
            if ((kk & himask) == prefix)
                atomicAdd(&hist[(unsigned)(kk >> (8 * pass)) & 255u], 1u);
        }
        __syncthreads();
        if (t == 0) {
            unsigned cum = 0;
            for (int b = 0; b < 256; ++b) {
                unsigned h = hist[b];
                if (cum + h >= sh_kth) {
                    sh_prefix |= (unsigned long long)b << (8 * pass);
                    sh_kth -= cum;
                    break;
                }
                cum += h;
            }
        }
        __syncthreads();
        prefix = sh_prefix;
    }
    const unsigned long long T = prefix;   // exact 100th-smallest key

    // ---- compact the 100 keys <= T ----
    if (t < 128) sel[t] = ~0ull;
    __syncthreads();
    for (int n = t; n < NN; n += 256) {
        unsigned long long kk = keys[n];
        if (kk <= T) {
            unsigned p = atomicAdd(&selCnt, 1u);
            if (p < 128) sel[p] = kk;
        }
    }
    __syncthreads();

    // ---- bitonic sort 128 keys ascending ----
    for (int ksz = 2; ksz <= 128; ksz <<= 1) {
        for (int j = ksz >> 1; j > 0; j >>= 1) {
            if (t < 128) {
                int i = t, ixj = i ^ j;
                if (ixj > i) {
                    bool up = ((i & ksz) == 0);
                    unsigned long long a = sel[i], b = sel[ixj];
                    if ((a > b) == up) { sel[i] = b; sel[ixj] = a; }
                }
            }
            __syncthreads();
        }
    }

    // ---- emit sorted dist (normalized), theta, mask, idx ----
    float norm = ord2f((unsigned)(sel[LL - 1] >> 32)) + 1e-6f;
    for (int r = t; r < LL; r += 256) {
        unsigned long long kk = sel[r];
        unsigned n = (unsigned)kk;
        sDist[row * LL + r]  = ord2f((unsigned)(kk >> 32)) / norm;
        sTheta[row * LL + r] = theta[base + n];
        sMask[row * LL + r]  = ninf[base + n];
        sIdx[row * LL + r]   = (int)n;
    }
}

// ---------------------------------------------------------------------------
// Kernel 3: per-row fused local attention with WMMA. One wave (32 lanes) per row.
//   S  = init_k[128,128] @ qk^T[128,16]   (8 M-tiles x 4 K-steps, wmma f16)
//   softmax over L -> wA[16,128]
//   U  = wA[16,128] @ init_k[128,128]     (8 N-tiles x 4 K-steps, wmma f16)
//   att/mh/score2 as short VALU dot loops; scatter into out.
// ---------------------------------------------------------------------------
__global__ __launch_bounds__(32) void attn_kernel(
    const float* __restrict__ sDist, const float* __restrict__ sTheta,
    const float* __restrict__ sMask, const int* __restrict__ sIdx,
    const float* __restrict__ qk,      // [8][128]
    const float* __restrict__ W_init,  // [128][2]
    const float* __restrict__ b_init,  // [128]
    const float* __restrict__ Wv,      // [128][128]
    const float* __restrict__ Wc,      // [128][128]
    const float* __restrict__ bc,      // [128]
    float* __restrict__ out) {
    __shared__ _Float16 initk[LP][EE];   // 32 KB  (rows >= LL are zero)
    __shared__ _Float16 qkB[16][LP];     //  4 KB  per-column contiguous B operand
    __shared__ _Float16 wA[16][LP];      //  4 KB  softmax weights as A operand
    __shared__ float    scoreS[LP][16];  //  8 KB
    __shared__ float    U[16][EE];       //  8 KB
    __shared__ float    attS[HH * DD];
    __shared__ float    mhS[EE];
    __shared__ float    maskS[LP];
    __shared__ float    stD[LL], stT[LL];

    const int row  = blockIdx.x;
    const int lane = threadIdx.x;
    const int m16  = lane & 15;
    const int sub  = lane >> 4;
    const float log_inc = 0.14624448352f;   // ln(10000)/63

    // ---- async-stage this row's top-k data into LDS (ASYNCcnt path) ----
    for (int l = lane; l < LL; l += 32) {
        async_g2l_b32(&stD[l],   &sDist[row * LL + l]);
        async_g2l_b32(&stT[l],   &sTheta[row * LL + l]);
        async_g2l_b32(&maskS[l], &sMask[row * LL + l]);
    }
    for (int l = LL + lane; l < LP; l += 32) maskS[l] = 0.0f;
    wait_async0();
    __syncthreads();

    // ---- build init_k (f16) + positional encoding; zero pad rows ----
    for (int t = lane; t < LP * EE; t += 32) {
        int l = t >> 7, e = t & 127;
        float v = 0.0f;
        if (l < LL) {
            float sd = stD[l];
            float st = stT[l];
            int   j  = (e < 64) ? e : (e - 64);
            float ang = (float)l * __expf(-(float)j * log_inc);
            float pos = (e < 64) ? __sinf(ang) : __cosf(ang);
            v = sd * W_init[2 * e + 0] + st * W_init[2 * e + 1] + b_init[e] + pos;
        }
        initk[l][e] = (_Float16)v;
    }
    for (int t = lane; t < 16 * LP; t += 32) {
        int n = t >> 7, k = t & 127;
        qkB[n][k] = (n < HH) ? (_Float16)qk[n * EE + k] : (_Float16)0.0f;
    }
    __syncthreads();

    // ---- S = init_k @ qk^T  (scores: rows=l, cols=h) ----
    for (int mt = 0; mt < 8; ++mt) {
        v8f c = {};
#pragma unroll
        for (int kt = 0; kt < 4; ++kt) {
            const int kb = kt * 32;
            v16h a, b;
#pragma unroll
            for (int i = 0; i < 8; ++i) {
                a[i]     = initk[mt * 16 + m16][kb + sub * 8 + i];       // K 0..7 / 8..15
                a[8 + i] = initk[mt * 16 + m16][kb + 16 + sub * 8 + i];  // K 16..23 / 24..31
            }
#pragma unroll
            for (int i = 0; i < 16; ++i)
                b[i] = qkB[m16][kb + (i < 8 ? sub * 8 + i : 16 + sub * 8 + (i - 8))]; // B[k][n], n = lane&15
            c = __builtin_amdgcn_wmma_f32_16x16x32_f16(
                    false, a, false, b, (short)0, c, false, false);
        }
#pragma unroll
        for (int v = 0; v < 8; ++v) {
            int l = mt * 16 + v + 8 * sub;
            float s = c[v] * 0.25f + maskS[l];                           // /sqrt(D=16) + mask
            scoreS[l][m16] = (l < LL) ? s : -__builtin_inff();
        }
    }
    __syncthreads();

    // ---- softmax over L per head; pack as A operand wA[h][l] ----
    if (lane < 16) {
        int h = lane;
        if (h < HH) {
            float mx = -__builtin_inff();
            for (int l = 0; l < LP; ++l) mx = fmaxf(mx, scoreS[l][h]);
            float sum = 0.0f;
            for (int l = 0; l < LP; ++l) sum += __expf(scoreS[l][h] - mx);
            float inv = 1.0f / sum;
            for (int l = 0; l < LP; ++l)
                wA[h][l] = (_Float16)(__expf(scoreS[l][h] - mx) * inv);
        } else {
            for (int l = 0; l < LP; ++l) wA[h][l] = (_Float16)0.0f;
        }
    }
    __syncthreads();

    // ---- U = w @ init_k  (U[h,e] = sum_l w[h,l]*init_k[l,e]) ----
    for (int nt = 0; nt < 8; ++nt) {
        v8f c = {};
#pragma unroll
        for (int kt = 0; kt < 4; ++kt) {
            const int kb = kt * 32;
            v16h a, b;
#pragma unroll
            for (int i = 0; i < 8; ++i) {
                a[i]     = wA[m16][kb + sub * 8 + i];
                a[8 + i] = wA[m16][kb + 16 + sub * 8 + i];
            }
#pragma unroll
            for (int i = 0; i < 16; ++i)
                b[i] = initk[kb + (i < 8 ? sub * 8 + i : 16 + sub * 8 + (i - 8))][nt * 16 + m16]; // B[k][n]
            c = __builtin_amdgcn_wmma_f32_16x16x32_f16(
                    false, a, false, b, (short)0, c, false, false);
        }
#pragma unroll
        for (int v = 0; v < 8; ++v)
            U[v + 8 * sub][nt * 16 + m16] = c[v];
    }
    __syncthreads();

    // ---- att[j] = U[j>>4,:] . Wv[j,:]   (j = h*16+d) ----
    for (int j = lane; j < HH * DD; j += 32) {
        const int h = j >> 4;
        float acc = 0.0f;
        for (int e = 0; e < EE; ++e) acc += U[h][e] * Wv[j * EE + e];
        attS[j] = acc;
    }
    __syncthreads();

    // ---- mh = att @ Wc^T + bc ----
    for (int ep = lane; ep < EE; ep += 32) {
        float acc = bc[ep];
        for (int j = 0; j < HH * DD; ++j) acc += attS[j] * Wc[ep * (HH * DD) + j];
        mhS[ep] = acc;
    }
    __syncthreads();

    // ---- score2[l] = (init_k[l,:] . mh)/sqrt(E); scatter by idx ----
    const float invsqE = 0.088388347648318447f;  // 1/sqrt(128)
    for (int l = lane; l < LL; l += 32) {
        float acc = 0.0f;
        for (int e = 0; e < EE; ++e) acc += (float)initk[l][e] * mhS[e];
        int n = sIdx[row * LL + l];
        out[(size_t)row * NN + n] = acc * invsqE;
    }
}

// ---------------------------------------------------------------------------
extern "C" void kernel_launch(void* const* d_in, const int* in_sizes, int n_in,
                              void* d_out, int out_size, void* d_ws, size_t ws_size,
                              hipStream_t stream) {
    const float* dist   = (const float*)d_in[0];
    const float* theta  = (const float*)d_in[1];
    const float* ninf   = (const float*)d_in[2];
    const float* W_init = (const float*)d_in[3];
    const float* b_init = (const float*)d_in[4];
    const float* cur    = (const float*)d_in[5];
    const float* Wq     = (const float*)d_in[6];
    const float* Wk     = (const float*)d_in[7];
    const float* Wv     = (const float*)d_in[8];
    const float* Wc     = (const float*)d_in[9];
    const float* bc     = (const float*)d_in[10];
    float* out = (float*)d_out;

    // workspace layout (floats)
    float* ws     = (float*)d_ws;
    float* qk     = ws;                      // 8*128
    float* sDist  = ws + 1024;               // ROWS*LL
    float* sTheta = sDist + ROWS * LL;       // ROWS*LL
    float* sMask  = sTheta + ROWS * LL;      // ROWS*LL
    int*   sIdx   = (int*)(sMask + ROWS * LL);

    zero_kernel<<<1024, 256, 0, stream>>>(out, ROWS * NN);
    prep_qk_kernel<<<1, 128, 0, stream>>>(Wq, Wk, cur, qk);
    topk_kernel<<<ROWS, 256, 0, stream>>>(dist, theta, ninf, sDist, sTheta, sMask, sIdx);
    attn_kernel<<<ROWS, 32, 0, stream>>>(sDist, sTheta, sMask, sIdx, qk,
                                         W_init, b_init, Wv, Wc, bc, out);
}
